// VanillaDynamicRouting_78477642432579
// MI455X (gfx1250) — matrix operand order repaired and checked by reading, not compiled, not run on hardware
//
#include <hip/hip_runtime.h>

// ---------------------------------------------------------------------------
// VanillaDynamicRouting on gfx1250:
//   u = x @ W^T  (16384x4096 * 4096x512) via v_wmma_f32_16x16x32_bf16
//   then 3 routing iterations (softmax / squash / agreement) per row.
//
// GEMM is compute-bound (AI ~230 FLOP/B); bf16 WMMA with f32 accumulation.
// B fragments staged through double-buffered LDS (1 barrier / K-step),
// all 4 B fragments loaded before the 4 back-to-back WMMAs.
// ---------------------------------------------------------------------------

typedef __attribute__((ext_vector_type(16))) __bf16 v16bf;
typedef __attribute__((ext_vector_type(8)))  float  v8f;
typedef __attribute__((ext_vector_type(4)))  float  v4f;

#define B_DIM 16384
#define D_DIM 4096
#define K_DIM 512

__device__ __forceinline__ void stage_b(const float* __restrict__ bPtr,
                                        __bf16* __restrict__ dst, int kk) {
    v4f w0 = *(const v4f*)(bPtr + kk);
    v4f w1 = *(const v4f*)(bPtr + kk + 4);
    dst[0] = (__bf16)w0.x; dst[1] = (__bf16)w0.y;
    dst[2] = (__bf16)w0.z; dst[3] = (__bf16)w0.w;
    dst[4] = (__bf16)w1.x; dst[5] = (__bf16)w1.y;
    dst[6] = (__bf16)w1.z; dst[7] = (__bf16)w1.w;
}

// ---------------------------------------------------------------------------
// GEMM: each 256-thread block (8 waves) computes a 128(M) x 64(N) tile of u.
// Wave w -> rows [mBase + 16w, +16); all waves share the same 64 N columns,
// so B fragments are converted f32->bf16 once per block per K-step.
// ---------------------------------------------------------------------------
__global__ __launch_bounds__(256) void routing_gemm_wmma(
    const float* __restrict__ x,   // [B_DIM, D_DIM]
    const float* __restrict__ W,   // [K_DIM, D_DIM]
    float* __restrict__ u)         // [B_DIM, K_DIM]
{
    // double-buffered: 2 x (4 nsub * 32 lanes) fragments of 16 bf16 = 8 KB
    __shared__ v16bf lbv[2][128];

    const int tid  = threadIdx.x;
    const int lane = tid & 31;
    const int wave = tid >> 5;

    const int mBase = blockIdx.y * 128 + wave * 16;
    const int nBase = blockIdx.x * 64;

    v8f acc[4];
#pragma unroll
    for (int i = 0; i < 4; ++i) acc[i] = (v8f){0.f,0.f,0.f,0.f,0.f,0.f,0.f,0.f};

    // --- A fragment addressing (16-bit A 16x32 layout):
    // lanes 0-15 : row M=lane,    K = 0..7  and 16..23
    // lanes 16-31: row M=lane-16, K = 8..15 and 24..31
    const int    mRow  = mBase + (lane & 15);
    const int    aHalf = (lane >> 4) << 3;          // 0 or 8
    const float* aPtr  = x + (size_t)mRow * D_DIM;

    // --- B staging: thread t fills half of fragment slot (t>>1).
    // slot = nsub*32 + blane ; fragment lane blane holds column
// n = nBase + nsub*16 + (blane&15), K = (blane<16 ? 0..7,16..23 : 8..15,24..31).
    const int    slot  = tid >> 1;                  // 0..127
    const int    half  = tid & 1;                   // 0/1 -> K +0 / +16
    const int    bn    = nBase + ((slot >> 5) << 4) + (slot & 15);
    const int    bk    = (((slot & 31) >> 4) << 3) + half * 16;
    const float* bPtr  = W + (size_t)bn * D_DIM + bk;
    const int    loff  = slot * 16 + half * 8;

    // prologue: stage K-step 0 into buffer 0
    stage_b(bPtr, ((__bf16*)lbv[0]) + loff, 0);
    __syncthreads();

    int buf = 0;
    for (int kk = 0; kk < D_DIM; kk += 32) {
        // ---- per-wave A fragment (two contiguous 8-float chunks)
        v4f a0 = *(const v4f*)(aPtr + kk + aHalf);
        v4f a1 = *(const v4f*)(aPtr + kk + aHalf + 4);
        v4f a2 = *(const v4f*)(aPtr + kk + aHalf + 16);
        v4f a3 = *(const v4f*)(aPtr + kk + aHalf + 20);
        // prefetch next K-step of the A stream (global_prefetch_b8)
        __builtin_prefetch(aPtr + kk + 32, 0, 1);

        // ---- stage next K-step into the other buffer (overlaps WMMA)
        if (kk + 32 < D_DIM)
            stage_b(bPtr, ((__bf16*)lbv[buf ^ 1]) + loff, kk + 32);

        v16bf aF;
        aF[0]  = (__bf16)a0.x; aF[1]  = (__bf16)a0.y;
        aF[2]  = (__bf16)a0.z; aF[3]  = (__bf16)a0.w;
        aF[4]  = (__bf16)a1.x; aF[5]  = (__bf16)a1.y;
        aF[6]  = (__bf16)a1.z; aF[7]  = (__bf16)a1.w;
        aF[8]  = (__bf16)a2.x; aF[9]  = (__bf16)a2.y;
        aF[10] = (__bf16)a2.z; aF[11] = (__bf16)a2.w;
        aF[12] = (__bf16)a3.x; aF[13] = (__bf16)a3.y;
        aF[14] = (__bf16)a3.z; aF[15] = (__bf16)a3.w;

        // ---- load all 4 B fragments, then issue 4 WMMAs back-to-back
        v16bf bF0 = lbv[buf][0 * 32 + lane];
        v16bf bF1 = lbv[buf][1 * 32 + lane];
        v16bf bF2 = lbv[buf][2 * 32 + lane];
        v16bf bF3 = lbv[buf][3 * 32 + lane];

        // (neg_a, A, neg_b, B, c_mod, C, reuse_a, reuse_b)
        acc[0] = __builtin_amdgcn_wmma_f32_16x16x32_bf16(
            false, aF, false, bF0, (short)0, acc[0], false, false);
        acc[1] = __builtin_amdgcn_wmma_f32_16x16x32_bf16(
            false, aF, false, bF1, (short)0, acc[1], false, false);
        acc[2] = __builtin_amdgcn_wmma_f32_16x16x32_bf16(
            false, aF, false, bF2, (short)0, acc[2], false, false);
        acc[3] = __builtin_amdgcn_wmma_f32_16x16x32_bf16(
            false, aF, false, bF3, (short)0, acc[3], false, false);

        __syncthreads();   // stage(next) complete + reads of buf complete
        buf ^= 1;
    }

    // ---- store: C/D layout: lanes 0-15 VGPR j -> (M=j, N=lane),
    //                         lanes 16-31 VGPR j -> (M=j+8, N=lane-16)
    const int colLane = lane & 15;
    const int rowOff  = (lane < 16) ? 0 : 8;
#pragma unroll
    for (int ns = 0; ns < 4; ++ns) {
        const int col = nBase + ns * 16 + colLane;
#pragma unroll
        for (int j = 0; j < 8; ++j) {
            const int row = mBase + rowOff + j;
            u[(size_t)row * K_DIM + col] = acc[ns][j];
        }
    }
}

// ---------------------------------------------------------------------------
// Routing: one block per row of u (K=512, 256 threads x 2 elements).
// ---------------------------------------------------------------------------
__device__ __forceinline__ float waveSum(float v) {
#pragma unroll
    for (int o = 16; o > 0; o >>= 1) v += __shfl_xor(v, o, 32);
    return v;
}
__device__ __forceinline__ float waveMax(float v) {
#pragma unroll
    for (int o = 16; o > 0; o >>= 1) v = fmaxf(v, __shfl_xor(v, o, 32));
    return v;
}
__device__ __forceinline__ float blockSum(float v, float* red) {
    const int lane = threadIdx.x & 31, wv = threadIdx.x >> 5;
    v = waveSum(v);
    __syncthreads();
    if (lane == 0) red[wv] = v;
    __syncthreads();
    float r = red[0];
#pragma unroll
    for (int i = 1; i < 8; ++i) r += red[i];
    return r;
}
__device__ __forceinline__ float blockMax(float v, float* red) {
    const int lane = threadIdx.x & 31, wv = threadIdx.x >> 5;
    v = waveMax(v);
    __syncthreads();
    if (lane == 0) red[wv] = v;
    __syncthreads();
    float r = red[0];
#pragma unroll
    for (int i = 1; i < 8; ++i) r = fmaxf(r, red[i]);
    return r;
}

__global__ __launch_bounds__(256) void routing_iterate(
    const float* __restrict__ u,   // [B_DIM, K_DIM]
    float* __restrict__ out)       // [B_DIM, K_DIM]
{
    __shared__ float red[8];
    const int row = blockIdx.x;
    const int tid = threadIdx.x;
    const float* ur = u + (size_t)row * K_DIM;

    const float u0 = ur[tid];
    const float u1 = ur[tid + 256];
    float b0 = 0.f, b1 = 0.f;

#pragma unroll
    for (int it = 0; it < 3; ++it) {
        // c = softmax(b)
        const float m  = blockMax(fmaxf(b0, b1), red);
        const float e0 = __expf(b0 - m), e1 = __expf(b1 - m);
        const float Zi = 1.0f / blockSum(e0 + e1, red);
        // s = u * c
        const float s0 = u0 * e0 * Zi, s1 = u1 * e1 * Zi;
        // v = squash(s)
        const float n2 = blockSum(s0 * s0 + s1 * s1, red);
        const float sc = n2 / (1.0f + n2) / (sqrtf(n2) + 1e-9f);
        // b += sum(u * v)
        const float a  = blockSum(u0 * s0 * sc + u1 * s1 * sc, red);
        b0 += a; b1 += a;
    }

    const float m  = blockMax(fmaxf(b0, b1), red);
    const float e0 = __expf(b0 - m), e1 = __expf(b1 - m);
    const float Zi = 1.0f / blockSum(e0 + e1, red);
    out[(size_t)row * K_DIM + tid]       = e0 * Zi;
    out[(size_t)row * K_DIM + tid + 256] = e1 * Zi;
}

// ---------------------------------------------------------------------------
extern "C" void kernel_launch(void* const* d_in, const int* in_sizes, int n_in,
                              void* d_out, int out_size, void* d_ws, size_t ws_size,
                              hipStream_t stream) {
    const float* x = (const float*)d_in[0];   // [16384, 4096]
    const float* W = (const float*)d_in[1];   // [512, 4096]
    float* out = (float*)d_out;               // [16384, 512]
    float* u   = (float*)d_ws;                // 32 MB scratch for logits

    dim3 gemmGrid(K_DIM / 64, B_DIM / 128);   // (8, 128)
    routing_gemm_wmma<<<gemmGrid, 256, 0, stream>>>(x, W, u);
    routing_iterate<<<B_DIM, 256, 0, stream>>>(u, out);
}
